// _OffsetPredictor_59021440582246
// MI455X (gfx1250) — matrix operand order repaired and checked by reading, not compiled, not run on hardware
//
#include <hip/hip_runtime.h>
#include <hip/hip_bf16.h>
#include <stdint.h>

typedef __attribute__((ext_vector_type(16))) __bf16 v16bf;
typedef __attribute__((ext_vector_type(8)))  float  v8f;

#define EPSV  1e-5f
#define SLOPE 0.01f

// Problem geometry (fixed by the reference) -- compile-time so all index math folds.
#define GB 2
#define GD 64
#define GH 64
#define GW 16
#define WSH 4      // log2(GW)
#define HSH 6      // log2(GH)
#define DSH 6      // log2(GD)
#define DHW (GD * GH * GW)   // 65536

__device__ __forceinline__ uint16_t f2bf(float f) {
  union { float f; uint32_t u; } c; c.f = f;
  uint32_t u = c.u;
  return (uint16_t)((u + 0x7FFFu + ((u >> 16) & 1u)) >> 16);  // RNE
}
__device__ __forceinline__ float bf2f(uint16_t h) {
  union { float f; uint32_t u; } c; c.u = ((uint32_t)h) << 16;
  return c.f;
}

// ---- CDNA5 async copy global->LDS (ASYNCcnt), see CDNA5 ISA
__device__ __forceinline__ void async_ld_b128(const uint16_t* g, uint32_t lds_off) {
  asm volatile("global_load_async_to_lds_b128 %0, %1, off"
               :: "v"(lds_off), "v"(g) : "memory");
}
__device__ __forceinline__ void async_ld_b64(const uint16_t* g, uint32_t lds_off) {
  asm volatile("global_load_async_to_lds_b64 %0, %1, off"
               :: "v"(lds_off), "v"(g) : "memory");
}
__device__ __forceinline__ void wait_async0() {
  asm volatile("s_wait_asynccnt 0" ::: "memory");
}
__device__ __forceinline__ uint32_t lds_off32(const void* p) {
  return (uint32_t)(uintptr_t)p;   // addr[31:0] = LDS byte offset (flat LDS aperture)
}

// ---------------------------------------------------------------------------
// fea: NCDHW fp32 -> channels-last bf16
// ---------------------------------------------------------------------------
__global__ void repack_fea(const float* __restrict__ src, uint16_t* __restrict__ dst) {
  size_t idx = (size_t)blockIdx.x * blockDim.x + threadIdx.x;
  const int C = 256;
  int c = (int)(idx & (C - 1));
  size_t m = idx >> 8;
  int w = (int)(m & (GW - 1));
  int h = (int)((m >> WSH) & (GH - 1));
  int d = (int)((m >> (WSH + HSH)) & (GD - 1));
  int b = (int)(m >> (WSH + HSH + DSH));
  size_t s = ((((size_t)b * C + c) * GD + d) * GH + h) * GW + w;
  dst[idx] = f2bf(src[s]);
}

// ---------------------------------------------------------------------------
// weights: OIDHW fp32 -> [Cout][K] bf16, K = 27*Cin, k = tap*Cin + ci
// ---------------------------------------------------------------------------
__global__ void repack_w(const float* __restrict__ src, uint16_t* __restrict__ dst,
                         int Cout, int Cin, int K) {
  int idx = blockIdx.x * blockDim.x + threadIdx.x;
  if (idx >= Cout * K) return;
  int k = idx % K, n = idx / K;
  int kidx = k / Cin, ci = k % Cin;
  dst[idx] = f2bf(src[((size_t)n * Cin + ci) * 27 + kidx]);
}

// ---------------------------------------------------------------------------
// Implicit-GEMM 3x3x3 SAME conv + fused BN + leaky ReLU, bf16 CL in/out.
// 256 thr = 8 waves (WM x WN), wave tile 32x32 (4 v_wmma per 32-K stage).
// Per tap: halo rows pre-zeroed once into both LDS ping-pong buffers; the
// Cin-chunk loop is pure async-to-LDS double buffering (one barrier/stage).
// Epilogue: BN+LReLU, staged through LDS for coalesced b128 stores.
// ---------------------------------------------------------------------------
template <int WM, int WN, int CIN, int COUT>
__global__ __launch_bounds__(256) void conv3d_wmma(
    const uint16_t* __restrict__ in,    // (M x CIN) bf16 channels-last
    const uint16_t* __restrict__ wk,    // (COUT x K) bf16
    uint16_t* __restrict__ out,         // (M x COUT) bf16 channels-last
    const float* __restrict__ bn_g, const float* __restrict__ bn_b,
    const float* __restrict__ bn_m, const float* __restrict__ bn_v) {
  constexpr int K        = 27 * CIN;
  constexpr int CHUNKS   = CIN / 32;         // Cin chunks per tap
  constexpr int M_TILE   = WM * 32;
  constexpr int N_TILE   = WN * 32;
  constexpr int PITCH    = 40;               // halves; 80-B rows, 16-B aligned vectors
  constexpr int A_HALVES = M_TILE * PITCH;
  constexpr int B_HALVES = N_TILE * PITCH;
  constexpr int RPT      = M_TILE / 64;      // A rows per thread (2 or 4)
  constexpr int EPITCH   = N_TILE + 8;       // epilogue pitch (bytes %16 == 0)

  __shared__ uint16_t a_tile[2 * A_HALVES];  // also reused as epilogue stage
  __shared__ uint16_t b_tile[2 * B_HALVES];

  const int t    = threadIdx.x;
  const int lane = t & 31;
  const int wv   = t >> 5;
  const int mi   = wv % WM;
  const int ni   = wv / WM;
  const int lmod = lane & 15;
  const int lhi  = lane >> 4;

  const int mBase = blockIdx.x * M_TILE;     // M = 131072 fits int
  const int nBase = blockIdx.y * N_TILE;
  const int b     = mBase >> (WSH + HSH + DSH);   // M_TILE divides DHW

  // A-loader geometry: thread t loads rows (t>>2)+64*j, 8-ch segment t&3
  const int aseg = t & 3;
  int dR[RPT], hR[RPT], wR[RPT], rA[RPT];
#pragma unroll
  for (int j = 0; j < RPT; ++j) {
    rA[j] = (t >> 2) + 64 * j;
    int m = mBase + rA[j];
    wR[j] = m & (GW - 1);
    hR[j] = (m >> WSH) & (GH - 1);
    dR[j] = (m >> (WSH + HSH)) & (GD - 1);
  }

  v8f acc[2][2];
#pragma unroll
  for (int i = 0; i < 2; ++i)
#pragma unroll
    for (int j = 0; j < 2; ++j)
      acc[i][j] = v8f{0.f, 0.f, 0.f, 0.f, 0.f, 0.f, 0.f, 0.f};

  int kbase = 0;   // tap * CIN
  for (int kd = 0; kd < 3; ++kd) {
    for (int kh = 0; kh < 3; ++kh) {
      for (int kw = 0; kw < 3; ++kw) {
        // ---- per-tap hoisted halo checks + row base pointers
        bool valA[RPT];
        const uint16_t* baseA[RPT];
#pragma unroll
        for (int j = 0; j < RPT; ++j) {
          const int dd = dR[j] + kd - 1, hh = hR[j] + kh - 1, ww = wR[j] + kw - 1;
          valA[j] = (unsigned)dd < GD && (unsigned)hh < GH && (unsigned)ww < GW;
          baseA[j] = in + ((((size_t)b * GD + dd) * GH + hh) * GW + ww) * CIN;
        }

        // Pre-zero halo rows ONCE per tap, in both ping-pong buffers.
        // Safe: previous tap's trailing barrier guarantees no readers; the
        // masked-off async loads below never touch these rows again this tap.
#pragma unroll
        for (int j = 0; j < RPT; ++j) {
          if (!valA[j]) {
            const uint4 z = make_uint4(0u, 0u, 0u, 0u);
            *(uint4*)(a_tile + rA[j] * PITCH + aseg * 8)            = z;
            *(uint4*)(a_tile + A_HALVES + rA[j] * PITCH + aseg * 8) = z;
          }
        }

        auto load_chunk = [&](int cc, int buf) {
          const int ci0 = cc * 32;
          uint16_t* ab = a_tile + buf * A_HALVES;
          uint16_t* bb = b_tile + buf * B_HALVES;
#pragma unroll
          for (int j = 0; j < RPT; ++j) {
            if (valA[j]) {
              async_ld_b128(baseA[j] + ci0 + aseg * 8,
                            lds_off32(ab + rA[j] * PITCH + aseg * 8));
            }
          }
          if constexpr (WN == 2) {       // 64 rows x 4 b128 segs = 256 thr
            const int nrow = t >> 2, seg = t & 3;
            async_ld_b128(wk + (size_t)(nBase + nrow) * K + kbase + ci0 + seg * 8,
                          lds_off32(bb + nrow * PITCH + seg * 8));
          } else {                       // 32 rows x 8 b64 segs = 256 thr
            const int nrow = t >> 3, seg = t & 7;
            async_ld_b64(wk + (size_t)(nBase + nrow) * K + kbase + ci0 + seg * 4,
                         lds_off32(bb + nrow * PITCH + seg * 4));
          }
        };

        // ---- double-buffered stages: one barrier per stage
        load_chunk(0, 0);
        wait_async0();
        __syncthreads();
        for (int cc = 0; cc < CHUNKS; ++cc) {
          const int cur = cc & 1;
          if (cc + 1 < CHUNKS) load_chunk(cc + 1, cur ^ 1);

          const uint16_t* ab = a_tile + cur * A_HALVES;
          const uint16_t* bb = b_tile + cur * B_HALVES;
          union { v16bf v; uint4 q[2]; } af[2], bfv[2];
#pragma unroll
          for (int i = 0; i < 2; ++i) {
            const uint16_t* ap = ab + (mi * 32 + i * 16 + lmod) * PITCH + (lhi ? 8 : 0);
            af[i].q[0] = *(const uint4*)(ap);
            af[i].q[1] = *(const uint4*)(ap + 16);
          }
#pragma unroll
          for (int j = 0; j < 2; ++j) {
            const uint16_t* bp = bb + (ni * 32 + j * 16 + lmod) * PITCH + lhi * 8;
            bfv[j].q[0] = *(const uint4*)(bp);
            bfv[j].q[1] = *(const uint4*)(bp + 16);
          }
#pragma unroll
          for (int i = 0; i < 2; ++i)
#pragma unroll
            for (int j = 0; j < 2; ++j)
              acc[i][j] = __builtin_amdgcn_wmma_f32_16x16x32_bf16(
                  false, af[i].v, false, bfv[j].v, (short)0, acc[i][j], false, false);

          wait_async0();
          __syncthreads();
        }
        kbase += CIN;
      }
    }
  }

  // ---- fused BN + leaky ReLU, staged through LDS for coalesced b128 stores
  uint16_t* st = a_tile;   // M_TILE * EPITCH halves <= 2*A_HALVES
#pragma unroll
  for (int j = 0; j < 2; ++j) {
    const int n = nBase + ni * 32 + j * 16 + lmod;
    const float sc = bn_g[n] * rsqrtf(bn_v[n] + EPSV);
    const float sh = bn_b[n] - bn_m[n] * sc;
#pragma unroll
    for (int i = 0; i < 2; ++i) {
      const int mr = mi * 32 + i * 16 + lhi * 8;
#pragma unroll
      for (int r = 0; r < 8; ++r) {
        float v = acc[i][j][r] * sc + sh;
        v = v > 0.f ? v : SLOPE * v;
        st[(mr + r) * EPITCH + ni * 32 + j * 16 + lmod] = f2bf(v);
      }
    }
  }
  __syncthreads();
  constexpr int VPR = N_TILE / 8;                 // b128 vectors per output row
  constexpr int NST = (M_TILE * N_TILE / 8) / 256;  // b128 stores per thread (4)
#pragma unroll
  for (int s = 0; s < NST; ++s) {
    const int e   = t + s * 256;
    const int row = e / VPR;
    const int col = e % VPR;
    const uint4 val = *(const uint4*)(st + row * EPITCH + col * 8);
    *(uint4*)(out + (size_t)(mBase + row) * COUT + nBase + col * 8) = val;
  }
}

// ---------------------------------------------------------------------------
// gather (32ch contiguous) + concat xyz + 35->32 MLP + BN + ReLU + 32->3 MLP
// ---------------------------------------------------------------------------
__global__ __launch_bounds__(256) void gather_mlp(
    const uint16_t* __restrict__ x3, const int* __restrict__ grid_ind,
    const float* __restrict__ xyz,
    const float* __restrict__ Wo1, const float* __restrict__ bo1,
    const float* __restrict__ go, const float* __restrict__ bo,
    const float* __restrict__ mo, const float* __restrict__ vo,
    const float* __restrict__ Wo2, const float* __restrict__ bo2,
    float* __restrict__ outp, int BN, int N) {
  int p = blockIdx.x * blockDim.x + threadIdx.x;
  if (p >= BN) return;
  int b = p / N;
  int d = grid_ind[p * 3 + 0];
  int h = grid_ind[p * 3 + 1];
  int w = grid_ind[p * 3 + 2];
  size_t m = (((size_t)b * GD + d) * GH + h) * GW + w;
  const uint16_t* src = x3 + m * 32;

  float hv[35];
#pragma unroll
  for (int c = 0; c < 32; ++c) hv[c] = bf2f(src[c]);
  hv[32] = xyz[p * 3 + 0];
  hv[33] = xyz[p * 3 + 1];
  hv[34] = xyz[p * 3 + 2];

  float a2[32];
#pragma unroll
  for (int o = 0; o < 32; ++o) a2[o] = bo1[o];
  for (int i = 0; i < 35; ++i) {
    const float x = hv[i];
#pragma unroll
    for (int o = 0; o < 32; ++o) a2[o] = fmaf(x, Wo1[i * 32 + o], a2[o]);
  }
  float o3[3] = {bo2[0], bo2[1], bo2[2]};
#pragma unroll
  for (int o = 0; o < 32; ++o) {
    const float sc = go[o] * rsqrtf(vo[o] + EPSV);
    float y = (a2[o] - mo[o]) * sc + bo[o];
    y = y > 0.f ? y : 0.f;
    o3[0] = fmaf(y, Wo2[o * 3 + 0], o3[0]);
    o3[1] = fmaf(y, Wo2[o * 3 + 1], o3[1]);
    o3[2] = fmaf(y, Wo2[o * 3 + 2], o3[2]);
  }
  outp[p * 3 + 0] = o3[0];
  outp[p * 3 + 1] = o3[1];
  outp[p * 3 + 2] = o3[2];
}

// ---------------------------------------------------------------------------
extern "C" void kernel_launch(void* const* d_in, const int* in_sizes, int n_in,
                              void* d_out, int out_size, void* d_ws, size_t ws_size,
                              hipStream_t stream) {
  const float* fea      = (const float*)d_in[0];
  const int*   grid_ind = (const int*)  d_in[1];
  const float* xyz      = (const float*)d_in[2];
  const float* W1 = (const float*)d_in[3];
  const float *g1 = (const float*)d_in[4],  *b1 = (const float*)d_in[5];
  const float *m1 = (const float*)d_in[6],  *v1 = (const float*)d_in[7];
  const float* W2 = (const float*)d_in[8];
  const float *g2 = (const float*)d_in[9],  *b2 = (const float*)d_in[10];
  const float *m2 = (const float*)d_in[11], *v2 = (const float*)d_in[12];
  const float* W3 = (const float*)d_in[13];
  const float *g3 = (const float*)d_in[14], *b3 = (const float*)d_in[15];
  const float *m3 = (const float*)d_in[16], *v3 = (const float*)d_in[17];
  const float* Wo1 = (const float*)d_in[18];
  const float* bo1 = (const float*)d_in[19];
  const float *go = (const float*)d_in[20], *bo = (const float*)d_in[21];
  const float *mo = (const float*)d_in[22], *vo = (const float*)d_in[23];
  const float* Wo2 = (const float*)d_in[24];
  const float* bo2 = (const float*)d_in[25];

  const int C = 256, C2 = 64, C3 = 32, N = 100000;
  const int K1 = 27 * C, K2 = 27 * C, K3 = 27 * C2;
  const size_t M = (size_t)GB * DHW;  // 131072

  char* ws = (char*)d_ws;
  uint16_t* feaCL = (uint16_t*)(ws);
  uint16_t* x1    = (uint16_t*)(ws + (size_t)67108864);
  uint16_t* x2    = (uint16_t*)(ws + (size_t)134217728);
  uint16_t* x3    = (uint16_t*)(ws + (size_t)150994944);
  uint16_t* w1k   = (uint16_t*)(ws + (size_t)159383552);
  uint16_t* w2k   = (uint16_t*)(ws + (size_t)162922496);
  uint16_t* w3k   = (uint16_t*)(ws + (size_t)163807232);

  {
    size_t tot = (size_t)GB * C * DHW;
    repack_fea<<<dim3((unsigned)(tot / 256)), 256, 0, stream>>>(fea, feaCL);
    repack_w<<<dim3((C  * K1 + 255) / 256), 256, 0, stream>>>(W1, w1k, C,  C,  K1);
    repack_w<<<dim3((C2 * K2 + 255) / 256), 256, 0, stream>>>(W2, w2k, C2, C,  K2);
    repack_w<<<dim3((C3 * K3 + 255) / 256), 256, 0, stream>>>(W3, w3k, C3, C2, K3);
  }

  // conv1: 256 -> 256, block tile 128x64
  conv3d_wmma<4, 2, 256, 256><<<dim3((unsigned)(M / 128), 256 / 64), 256, 0, stream>>>(
      feaCL, w1k, x1, g1, b1, m1, v1);
  // conv2: 256 -> 64, block tile 128x64
  conv3d_wmma<4, 2, 256, 64><<<dim3((unsigned)(M / 128), 1), 256, 0, stream>>>(
      x1, w2k, x2, g2, b2, m2, v2);
  // conv3: 64 -> 32, block tile 256x32
  conv3d_wmma<8, 1, 64, 32><<<dim3((unsigned)(M / 256), 1), 256, 0, stream>>>(
      x2, w3k, x3, g3, b3, m3, v3);

  const int BN = GB * N;
  gather_mlp<<<dim3((BN + 255) / 256), 256, 0, stream>>>(
      x3, grid_ind, xyz, Wo1, bo1, go, bo, mo, vo, Wo2, bo2,
      (float*)d_out, BN, N);
}
